// Net_75118978007716
// MI455X (gfx1250) — matrix operand order repaired and checked by reading, not compiled, not run on hardware
//
#include <hip/hip_runtime.h>
#include <math.h>

// ---------------- problem constants ----------------
constexpr int Bc = 8, Tc = 512, Dc = 80, HDIMc = 512, CDIMc = 128;
constexpr int PADc = Dc - 1;          // 79
constexpr int NSc  = Dc + PADc;       // 159 shifts
constexpr int BTc  = Bc * Tc;         // 4096 rows
constexpr int KPAD = 96;              // D padded to multiple of 32
constexpr size_t N_BTD = (size_t)BTc * Dc;      // 327680
constexpr size_t N_BTH = (size_t)BTc * HDIMc;   // 2097152

typedef __attribute__((ext_vector_type(16))) _Float16 v16h;
typedef __attribute__((ext_vector_type(8)))  _Float16 v8h;
typedef __attribute__((ext_vector_type(8)))  float    v8f;

// ---------------- init ----------------
__global__ void init_copy_kernel(const float* __restrict__ x, const float* __restrict__ y,
                                 float* __restrict__ xres, float* __restrict__ yres) {
  for (size_t i = (size_t)blockIdx.x * blockDim.x + threadIdx.x; i < N_BTD;
       i += (size_t)gridDim.x * blockDim.x) {
    xres[i] = x[i];
    yres[i] = y[i];
  }
}

__global__ void init_mask_kernel(float* __restrict__ mprev) {
  for (size_t i = (size_t)blockIdx.x * blockDim.x + threadIdx.x; i < N_BTH;
       i += (size_t)gridDim.x * blockDim.x)
    mprev[i] = 0.f;
}

__global__ void init_row_kernel(const float* __restrict__ y, int* __restrict__ seqh,
                                float* __restrict__ loss) {
  int i = blockIdx.x * blockDim.x + threadIdx.x;
  if (i < 4) loss[i] = 0.f;
  if (i < BTc) {
    int all0 = 1;
    for (int j = 0; j < Dc; ++j)
      if (y[(size_t)i * Dc + j] != 0.f) { all0 = 0; break; }
    seqh[i] = all0;
  }
}

// W_enc [512][80] -> f16 K-major [96][512], zero-padded K
__global__ void conv_wenc_kernel(const float* __restrict__ W, _Float16* __restrict__ WT) {
  int i = blockIdx.x * blockDim.x + threadIdx.x;           // 96*512 threads
  if (i >= KPAD * HDIMc) return;
  int k = i / HDIMc, n = i % HDIMc;
  WT[i] = (k < Dc) ? (_Float16)W[(size_t)n * Dc + k] : (_Float16)0.f;
}

// W_dec [80][512] -> f16 K-major [512][80]
__global__ void conv_wdec_kernel(const float* __restrict__ W, _Float16* __restrict__ WT) {
  int i = blockIdx.x * blockDim.x + threadIdx.x;           // 512*80 threads
  if (i >= HDIMc * Dc) return;
  int k = i / Dc, n = i % Dc;
  WT[i] = (_Float16)W[(size_t)n * HDIMc + k];
}

// ---------------- selector + attention + x_ele (writes f16 padded y_att) ----------------
__global__ void selector_kernel(const float* __restrict__ xres, const float* __restrict__ yres,
                                _Float16* __restrict__ yatt16, float* __restrict__ xele,
                                int* __restrict__ thetaOut) {
  __shared__ float xs[Dc], ys[Dc], ya[Dc], xo[Dc], wv[Dc];
  __shared__ float sim[NSc];
  __shared__ float ynorm, wsum;
  __shared__ int thS;
  const int tid = threadIdx.x;
  const int row = blockIdx.x;
  const size_t base = (size_t)row * Dc;

  for (int j = tid; j < Dc; j += blockDim.x) { xs[j] = xres[base + j]; ys[j] = yres[base + j]; }
  __syncthreads();
  if (tid == 0) {
    float s = 0.f;
    for (int j = 0; j < Dc; ++j) s += ys[j] * ys[j];
    ynorm = sqrtf(s);
  }
  __syncthreads();
  for (int s = tid; s < NSc; s += blockDim.x) {   // x_aug[s][j] = x[s+j-PAD]
    float dot = 0.f, nx = 0.f;
    for (int j = 0; j < Dc; ++j) {
      int idx = s + j - PADc;
      if (idx >= 0 && idx < Dc) { float v = xs[idx]; dot += v * ys[j]; nx += v * v; }
    }
    sim[s] = dot / (sqrtf(nx) * ynorm + 1e-8f);
  }
  __syncthreads();
  if (tid == 0) {            // first-occurrence argmax (jnp.argmax semantics)
    float best = sim[0]; int bi = 0;
    for (int s = 1; s < NSc; ++s) if (sim[s] > best) { best = sim[s]; bi = s; }
    thS = bi;
  }
  __syncthreads();
  const int th = thS;
  if (tid < Dc) {
    int idx = th + tid - PADc;
    float x = (idx >= 0 && idx < Dc) ? xs[idx] : 0.f;
    xo[tid] = x;
    float sc = fmaxf(x * ys[tid], 1e-6f);
    wv[tid] = powf(sc, 0.1f);   // softmax(log(sc)/10) == sc^{0.1}/sum
  }
  __syncthreads();
  if (tid == 0) {
    float s = 0.f;
    for (int j = 0; j < Dc; ++j) s += wv[j];
    wsum = s;
  }
  __syncthreads();
  if (tid < Dc) ya[tid] = xo[tid] * (wv[tid] / wsum);
  __syncthreads();
  if (tid < KPAD)
    yatt16[(size_t)row * KPAD + tid] = (tid < Dc) ? (_Float16)ya[tid] : (_Float16)0.f;
  if (tid < Dc) {           // reverse_pad_for_shift: x_ele[j] = y_att[j + PAD - theta]
    int src = tid + PADc - th;
    xele[base + tid] = (src >= 0 && src < Dc) ? ya[src] : 0.f;
  }
  if (tid == 0) thetaOut[row] = th;
}

// ---------------- relay cummax (src_t) ----------------
__global__ void srct_kernel(const int* __restrict__ theta, int* __restrict__ srct) {
  int b = threadIdx.x;
  if (b >= Bc) return;
  int cm = -1;
  for (int t = 0; t < Tc; ++t) {
    int sh = theta[b * Tc + t] - (Dc - 1);
    if (sh < 0) sh = -sh;
    if (sh > 5) cm = t;                 // |theta-79| > ENERGY_TH(5.0)
    srct[b * Tc + t] = cm < 0 ? 0 : cm;
  }
}

// ---- fragment helpers: two contiguous 8-halfword chunks -> v16h ----
__device__ __forceinline__ v16h load_a_frag(const _Float16* rowp, int kk, int hi) {
  // A 16x32 fragment (ISA 7.12.2): lane group hi, octets at kk+8*hi and kk+16+8*hi
  const _Float16* p = rowp + kk + (hi << 3);
  v8h a0 = *(const v8h*)(p);
  v8h a1 = *(const v8h*)(p + 16);
  return __builtin_shufflevector(a0, a1, 0, 1, 2, 3, 4, 5, 6, 7, 8, 9, 10, 11, 12, 13, 14, 15);
}
__device__ __forceinline__ v16h load_b_frag(const _Float16* p, int ld, int hi) {
  // B fragment: lane = N column, K octets at 8*hi and 16+8*hi (row stride ld)
  v16h b;
#pragma unroll
  for (int s = 0; s < 8; ++s) {
    b[s]     = p[(size_t)((hi << 3) + s) * ld];
    b[s + 8] = p[(size_t)(16 + (hi << 3) + s) * ld];
  }
  return b;
}

// ---------------- GEMM h = y_att @ W_enc^T + b_enc (WMMA f16->f32) ----------------
__global__ void gemm_enc_wmma(const _Float16* __restrict__ A16,   // [BT][96]
                              const _Float16* __restrict__ BT16,  // [96][512] K-major
                              const float* __restrict__ bias,     // [HDIM]
                              float* __restrict__ Hout) {         // [BT][HDIM]
  const int lane = threadIdx.x & 31;
  const int wv   = threadIdx.x >> 5;
  const int tile = blockIdx.x * 8 + wv;   // 8192 tiles = 256 (M) x 32 (N)
  const int m0 = (tile >> 5) << 4;
  const int n0 = (tile & 31) << 4;
  const int r  = lane & 15;
  const int hi = lane >> 4;
  const _Float16* arow = A16 + (size_t)(m0 + r) * KPAD;
  v8f c = {};
#pragma unroll
  for (int kk = 0; kk < KPAD; kk += 32) {
    v16h a  = load_a_frag(arow, kk, hi);
    v16h bm = load_b_frag(BT16 + (size_t)kk * HDIMc + n0 + r, HDIMc, hi);
    c = __builtin_amdgcn_wmma_f32_16x16x32_f16(false, a, false, bm, (short)0, c, false, false);
  }
#pragma unroll
  for (int v = 0; v < 8; ++v) {           // C/D: VGPR v -> M = v + 8*hi, N = lane&15
    int m = m0 + v + (hi << 3);
    int n = n0 + r;
    Hout[(size_t)m * HDIMc + n] = c[v] + bias[n];
  }
}

// ---------------- GEMM y_ele = h @ W_dec^T + b_dec (async LDS B + WMMA) ----------------
__global__ void gemm_dec_wmma(const _Float16* __restrict__ A16,   // [BT][512]
                              const _Float16* __restrict__ BT16,  // [512][80] K-major
                              const float* __restrict__ bias,     // [D]
                              float* __restrict__ Yele) {         // [BT][D]
  __shared__ __align__(16) _Float16 Blds[HDIMc * Dc];             // 80 KB
  // Cooperative async stage of entire B matrix into LDS (CDNA5 async path).
  {
    unsigned ldsBase = (unsigned)(size_t)(void*)&Blds[0];
    const char* g = (const char*)BT16;
    constexpr int nChunks = (HDIMc * Dc * 2) / 16;                // 5120 x 16B
    for (int p = threadIdx.x; p < nChunks; p += blockDim.x) {
      unsigned off = (unsigned)p * 16u;
      asm volatile("global_load_async_to_lds_b128 %0, %1, off"
                   :: "v"(ldsBase + off), "v"(g + off) : "memory");
    }
    asm volatile("s_wait_asynccnt 0" ::: "memory");
    __syncthreads();
  }
  const int lane = threadIdx.x & 31;
  const int wv   = threadIdx.x >> 5;
  const int tile = blockIdx.x * 8 + wv;   // 1280 tiles = 256 (M) x 5 (N)
  const int m0 = (tile / 5) << 4;
  const int n0 = (tile % 5) << 4;
  const int r  = lane & 15;
  const int hi = lane >> 4;
  const _Float16* arow = A16 + (size_t)(m0 + r) * HDIMc;
  v8f c = {};
#pragma unroll 4
  for (int kk = 0; kk < HDIMc; kk += 32) {
    v16h a  = load_a_frag(arow, kk, hi);
    v16h bm = load_b_frag(&Blds[kk * Dc + n0 + r], Dc, hi);
    c = __builtin_amdgcn_wmma_f32_16x16x32_f16(false, a, false, bm, (short)0, c, false, false);
  }
#pragma unroll
  for (int v = 0; v < 8; ++v) {
    int m = m0 + v + (hi << 3);
    int n = n0 + r;
    Yele[(size_t)m * Dc + n] = c[v] + bias[n];
  }
}

// ---------------- exact top-128 membership (stable tie-break like lax.top_k) ----------------
__global__ void topk_kernel(const float* __restrict__ Hin, float* __restrict__ selm) {
  __shared__ float v[HDIMc];
  const int tid = threadIdx.x;
  const size_t base = (size_t)blockIdx.x * HDIMc;
  for (int j = tid; j < HDIMc; j += blockDim.x) { float h = Hin[base + j]; v[j] = h * h; }
  __syncthreads();
  for (int j = tid; j < HDIMc; j += blockDim.x) {
    float vj = v[j]; int rank = 0;
    for (int k = 0; k < HDIMc; ++k) {
      float vk = v[k];
      rank += (vk > vj) || (vk == vj && k < j);
    }
    selm[base + j] = (rank < CDIMc) ? 1.f : 0.f;
  }
}

// ---------------- hsr loss (iters >= 1): loss_h, then zero h where mask_prev>0 ----------------
__global__ void hsr_loss_kernel(float* __restrict__ H, const float* __restrict__ mprev,
                                const float* __restrict__ selm, const int* __restrict__ srct,
                                const int* __restrict__ seqh, float* __restrict__ loss, int iter) {
  __shared__ float red[256];
  float acc = 0.f;
  for (size_t i = (size_t)blockIdx.x * blockDim.x + threadIdx.x; i < N_BTH;
       i += (size_t)gridDim.x * blockDim.x) {
    int rowbt = (int)(i >> 9);
    int j = (int)(i & (HDIMc - 1));
    int b = rowbt >> 9;                       // T == 512
    int st = srct[rowbt];
    float mc0 = selm[((size_t)(b * Tc + st) << 9) + j];
    float mp  = mprev[i];
    float inter = mp * mc0;
    if (inter > 0.f && seqh[rowbt] == 0) {
      float d = H[i] - (1.f - inter);
      acc += d * d;
    }
    if (mp > 0.f) H[i] = 0.f;
  }
  red[threadIdx.x] = acc; __syncthreads();
  for (int s = 128; s > 0; s >>= 1) {
    if (threadIdx.x < s) red[threadIdx.x] += red[threadIdx.x + s];
    __syncthreads();
  }
  if (threadIdx.x == 0) atomicAdd(&loss[iter], red[0]);
}

// ---- mask_cur gather + mask_prev += ; emit masked h as f16 for the decode GEMM ----
__global__ void apply_mask_kernel(const float* __restrict__ H, float* __restrict__ mprev,
                                  const float* __restrict__ selm, const int* __restrict__ srct,
                                  _Float16* __restrict__ H16) {
  for (size_t i = (size_t)blockIdx.x * blockDim.x + threadIdx.x; i < N_BTH;
       i += (size_t)gridDim.x * blockDim.x) {
    int rowbt = (int)(i >> 9);
    int j = (int)(i & (HDIMc - 1));
    int b = rowbt >> 9;
    int st = srct[rowbt];
    float mc = selm[((size_t)(b * Tc + st) << 9) + j];
    mprev[i] += mc;
    H16[i] = (_Float16)((mc > 0.f) ? H[i] : 0.f);
  }
}

// ---------------- energy loss + residual updates ----------------
__global__ void energy_kernel(const float* __restrict__ yele, float* __restrict__ yres,
                              float* __restrict__ xres, const float* __restrict__ xele,
                              const int* __restrict__ theta, const float* __restrict__ yorig,
                              float* __restrict__ loss, int iter) {
  __shared__ float red[256];
  float acc = 0.f;
  for (size_t i = (size_t)blockIdx.x * blockDim.x + threadIdx.x; i < N_BTD;
       i += (size_t)gridDim.x * blockDim.x) {
    int rowbt = (int)(i / Dc);
    float sh = fabsf((float)(theta[rowbt] - (Dc - 1)));
    float ye = yele[i], yr = yres[i];
    float d = ye - yr;
    float ll = d * d;
    if (yorig[i] == 0.f) ll = 0.f;   // seq_mask
    if (sh > 5.0f) ll = 0.f;         // ENERGY_TH gate
    ll /= (sh + 1.f);
    acc += ll;
    yres[i] = yr - ye;
    xres[i] -= xele[i];
  }
  red[threadIdx.x] = acc; __syncthreads();
  for (int s = 128; s > 0; s >>= 1) {
    if (threadIdx.x < s) red[threadIdx.x] += red[threadIdx.x + s];
    __syncthreads();
  }
  if (threadIdx.x == 0) atomicAdd(&loss[iter], red[0]);
}

__global__ void final_kernel(const float* __restrict__ loss, float* __restrict__ out) {
  if (threadIdx.x == 0 && blockIdx.x == 0)
    out[0] = (loss[0] + loss[1] + loss[2] + loss[3]) * 0.25f;
}

// ---------------- host orchestration ----------------
extern "C" void kernel_launch(void* const* d_in, const int* in_sizes, int n_in,
                              void* d_out, int out_size, void* d_ws, size_t ws_size,
                              hipStream_t stream) {
  const float* x    = (const float*)d_in[0];
  const float* y    = (const float*)d_in[1];
  const float* Wenc = (const float*)d_in[2];   // [512][80]
  const float* benc = (const float*)d_in[3];   // [512]
  const float* Wdec = (const float*)d_in[4];   // [80][512]
  const float* bdec = (const float*)d_in[5];   // [80]
  float* out = (float*)d_out;

  char* cur = (char*)d_ws;
  auto alloc = [&](size_t bytes) -> char* {
    char* p = cur;
    cur += (bytes + 255) & ~(size_t)255;
    return p;
  };
  float*     x_res   = (float*)alloc(N_BTD * 4);
  float*     y_res   = (float*)alloc(N_BTD * 4);
  float*     x_ele   = (float*)alloc(N_BTD * 4);
  float*     y_ele   = (float*)alloc(N_BTD * 4);
  float*     Hbuf    = (float*)alloc(N_BTH * 4);
  float*     selm    = (float*)alloc(N_BTH * 4);
  float*     mprev   = (float*)alloc(N_BTH * 4);
  _Float16*  yatt16  = (_Float16*)alloc((size_t)BTc * KPAD * 2);
  _Float16*  H16     = (_Float16*)alloc(N_BTH * 2);
  _Float16*  WencT16 = (_Float16*)alloc((size_t)KPAD * HDIMc * 2);
  _Float16*  WdecT16 = (_Float16*)alloc((size_t)HDIMc * Dc * 2);
  int*       theta   = (int*)alloc(BTc * 4);
  int*       srct    = (int*)alloc(BTc * 4);
  int*       seqh    = (int*)alloc(BTc * 4);
  float*     lossv   = (float*)alloc(8 * 4);

  init_copy_kernel<<<1280, 256, 0, stream>>>(x, y, x_res, y_res);
  init_mask_kernel<<<2048, 256, 0, stream>>>(mprev);
  init_row_kernel<<<(BTc + 255) / 256, 256, 0, stream>>>(y, seqh, lossv);
  conv_wenc_kernel<<<(KPAD * HDIMc + 255) / 256, 256, 0, stream>>>(Wenc, WencT16);
  conv_wdec_kernel<<<(HDIMc * Dc + 255) / 256, 256, 0, stream>>>(Wdec, WdecT16);

  for (int it = 0; it < 4; ++it) {
    selector_kernel<<<BTc, 192, 0, stream>>>(x_res, y_res, yatt16, x_ele, theta);
    srct_kernel<<<1, 32, 0, stream>>>(theta, srct);
    gemm_enc_wmma<<<1024, 256, 0, stream>>>(yatt16, WencT16, benc, Hbuf);
    topk_kernel<<<BTc, 256, 0, stream>>>(Hbuf, selm);
    if (it > 0) {
      hsr_loss_kernel<<<1024, 256, 0, stream>>>(Hbuf, mprev, selm, srct, seqh, lossv, it);
      topk_kernel<<<BTc, 256, 0, stream>>>(Hbuf, selm);   // relay on zeroed h
    }
    apply_mask_kernel<<<1024, 256, 0, stream>>>(Hbuf, mprev, selm, srct, H16);
    gemm_dec_wmma<<<160, 256, 0, stream>>>(H16, WdecT16, bdec, y_ele);
    energy_kernel<<<1024, 256, 0, stream>>>(y_ele, y_res, x_res, x_ele, theta, y, lossv, it);
  }
  final_kernel<<<1, 1, 0, stream>>>(lossv, out);
}
